// Encoder_76046690943021
// MI455X (gfx1250) — matrix operand order repaired and checked by reading, not compiled, not run on hardware
//
#include <hip/hip_runtime.h>
#include <hip/hip_bf16.h>
#include <stdint.h>

#define T_OBS 8
#define T_PRE 12
#define BATCH 4096
#define HDIM  1024
#define GDIM  4096   // 4*H

typedef __attribute__((ext_vector_type(16))) __bf16 v16bf;
typedef __attribute__((ext_vector_type(8)))  float  v8f;

__device__ __forceinline__ unsigned short f2bf(float f) {
  unsigned int u = __float_as_uint(f);
  u += 0x7FFFu + ((u >> 16) & 1u);     // round-to-nearest-even
  return (unsigned short)(u >> 16);
}
__device__ __forceinline__ float sigf(float x)     { return 1.0f / (1.0f + __expf(-x)); }
__device__ __forceinline__ float tanhfast(float x) { return 1.0f - 2.0f / (__expf(2.0f * x) + 1.0f); }

union FragBF {
  uint4 q[2];
  v16bf v;
};

// One LSTM step: gates = h@Whh^T (bf16 WMMA) + x@Wc^T + bc, then cell update.
// Block tile: 128 batch rows x 64 hidden cols, all 4 gate blocks computed here.
// Double-buffered LDS pipeline: global loads for chunk k+1 fly during WMMAs on chunk k.
__global__ __launch_bounds__(256, 1)
void lstm_step_kernel(const unsigned short* __restrict__ h_in,   // [B,H] bf16
                      unsigned short* __restrict__ h_out_bf,     // [B,H] bf16
                      float* __restrict__ h_out_f32,             // [B,H] f32
                      float* __restrict__ c_buf,                 // [B,H] f32, in-place
                      const unsigned short* __restrict__ whh,    // [4H,H] bf16
                      const float* __restrict__ wc,              // [4H,2] composed input weights
                      const float* __restrict__ bc,              // [4H] composed bias
                      const float* __restrict__ xt)              // [B,2]
{
  __shared__ __align__(16) unsigned short sA[2][128 * 40];  // 128 rows x 32 K, stride 40
  __shared__ __align__(16) unsigned short sB[2][256 * 40];  // 4 gates x 64 cols x 32 K
  __shared__ float xs[256];                                 // 128 rows x 2 feats

  const int tid = threadIdx.x;
  const int b0 = blockIdx.x * 128;
  const int n0 = blockIdx.y * 64;

  xs[tid] = xt[(size_t)b0 * 2 + tid];

  const int wid  = tid >> 5;
  const int lane = tid & 31;
  const int wm = wid & 1;        // which 64-row half
  const int wn = wid >> 1;       // which 16-col group (0..3)
  const int lr = lane & 15;      // M (A) / N (B) within 16
  const int lh = lane >> 4;      // K-half selector per ISA 16-bit layout
  const int kA = lh * 8;         // half-element offset within 32-wide K chunk

  v8f acc[4][4] = {};            // [gate][m-subtile], 16x16 f32 each

  const int arow = tid >> 1, ahf = tid & 1;
  const int bgate = tid >> 6, bcol = tid & 63;
  const size_t a_src_row = (size_t)(b0 + arow) * HDIM;
  const size_t b_src_row = (size_t)(bgate * HDIM + n0 + bcol) * HDIM;

  // Prologue: stage K-chunk 0 into buffer 0.
  {
    uint4 ra = *(const uint4*)(h_in + a_src_row + ahf * 16);
    uint4 ra1 = *(const uint4*)(h_in + a_src_row + ahf * 16 + 8);
    const uint4* s = (const uint4*)(whh + b_src_row);
    uint4 rb0 = s[0], rb1 = s[1], rb2 = s[2], rb3 = s[3];
    *(uint4*)(&sA[0][arow * 40 + ahf * 16]) = ra;
    *(uint4*)(&sA[0][arow * 40 + ahf * 16 + 8]) = ra1;
    *(uint4*)(&sB[0][tid * 40])             = rb0;
    *(uint4*)(&sB[0][tid * 40 + 8])         = rb1;
    *(uint4*)(&sB[0][tid * 40 + 16])        = rb2;
    *(uint4*)(&sB[0][tid * 40 + 24])        = rb3;
  }
  __syncthreads();

  int buf = 0;
  for (int k0 = 0; k0 < HDIM; k0 += 32, buf ^= 1) {
    const int kn = k0 + 32;
    const bool more = (kn < HDIM);   // block-uniform

    // Issue global loads for the NEXT chunk; they complete during the WMMAs below.
    uint4 na, na1, nb0, nb1, nb2, nb3;
    if (more) {
      na = *(const uint4*)(h_in + a_src_row + kn + ahf * 16);
      na1 = *(const uint4*)(h_in + a_src_row + kn + ahf * 16 + 8);
      const uint4* sn = (const uint4*)(whh + b_src_row + kn);
      nb0 = sn[0]; nb1 = sn[1]; nb2 = sn[2]; nb3 = sn[3];
      if (kn + 32 < HDIM) {
        __builtin_prefetch(h_in + a_src_row + kn + 32, 0, 1);
        __builtin_prefetch(whh + b_src_row + kn + 32, 0, 1);
      }
    }

    // Compute on current buffer.
    FragBF af[4];
    #pragma unroll
    for (int mi = 0; mi < 4; ++mi) {
      int r = wm * 64 + mi * 16 + lr;
      af[mi].q[0] = *(const uint4*)(&sA[buf][r * 40 + kA]);
      af[mi].q[1] = *(const uint4*)(&sA[buf][r * 40 + kA + 16]);
    }
    FragBF bfr[4];
    #pragma unroll
    for (int g = 0; g < 4; ++g) {
      int r = g * 64 + wn * 16 + lr;
      bfr[g].q[0] = *(const uint4*)(&sB[buf][r * 40 + kA]);
      bfr[g].q[1] = *(const uint4*)(&sB[buf][r * 40 + kA + 16]);
    }
    #pragma unroll
    for (int g = 0; g < 4; ++g)
      #pragma unroll
      for (int mi = 0; mi < 4; ++mi)
        acc[g][mi] = __builtin_amdgcn_wmma_f32_16x16x32_bf16(
            false, af[mi].v, false, bfr[g].v, (short)0, acc[g][mi], false, false);

    // Drain the in-flight global loads into the other buffer.
    if (more) {
      const int nb = buf ^ 1;
      *(uint4*)(&sA[nb][arow * 40 + ahf * 16]) = na;
      *(uint4*)(&sA[nb][arow * 40 + ahf * 16 + 8]) = na1;
      *(uint4*)(&sB[nb][tid * 40])             = nb0;
      *(uint4*)(&sB[nb][tid * 40 + 8])         = nb1;
      *(uint4*)(&sB[nb][tid * 40 + 16])        = nb2;
      *(uint4*)(&sB[nb][tid * 40 + 24])        = nb3;
    }
    __syncthreads();   // one barrier per iteration covers both hazards
  }

  // Epilogue: fused input projection + bias + LSTM cell update.
  const int col = n0 + wn * 16 + lr;
  float wc0[4], wc1[4], bcv[4];
  #pragma unroll
  for (int g = 0; g < 4; ++g) {
    int gc = g * HDIM + col;
    wc0[g] = wc[gc * 2];
    wc1[g] = wc[gc * 2 + 1];
    bcv[g] = bc[gc];
  }
  #pragma unroll
  for (int mi = 0; mi < 4; ++mi) {
    #pragma unroll
    for (int v = 0; v < 8; ++v) {
      int rloc = wm * 64 + mi * 16 + lh * 8 + v;   // C layout: lanes 16-31 -> M+8
      float x0 = xs[rloc * 2], x1 = xs[rloc * 2 + 1];
      float vi = acc[0][mi][v] + x0 * wc0[0] + x1 * wc1[0] + bcv[0];
      float vf = acc[1][mi][v] + x0 * wc0[1] + x1 * wc1[1] + bcv[1];
      float vg = acc[2][mi][v] + x0 * wc0[2] + x1 * wc1[2] + bcv[2];
      float vo = acc[3][mi][v] + x0 * wc0[3] + x1 * wc1[3] + bcv[3];
      size_t idx = (size_t)(b0 + rloc) * HDIM + col;
      float cn = sigf(vf) * c_buf[idx] + sigf(vi) * tanhfast(vg);
      float hn = sigf(vo) * tanhfast(cn);
      c_buf[idx]     = cn;
      h_out_f32[idx] = hn;
      h_out_bf[idx]  = f2bf(hn);
    }
  }
}

__global__ void cvt_bf16_kernel(const float* __restrict__ src,
                                unsigned short* __restrict__ dst, int n) {
  int i = blockIdx.x * blockDim.x + threadIdx.x;
  if (i < n) dst[i] = f2bf(src[i]);
}

// Wc = W_ih @ W_in (4Hx2), bc = b + W_ih @ b_in  (folds the input-embedding GEMM)
__global__ void compose_kernel(const float* __restrict__ W_ih, const float* __restrict__ W_in,
                               const float* __restrict__ b_in, const float* __restrict__ bsum,
                               float* __restrict__ wc, float* __restrict__ bc) {
  int g = blockIdx.x * blockDim.x + threadIdx.x;
  if (g >= GDIM) return;
  float a0 = 0.f, a1 = 0.f, ab = 0.f;
  for (int k = 0; k < 256; ++k) {
    float w = W_ih[(size_t)g * 256 + k];
    a0 += w * W_in[k * 2];
    a1 += w * W_in[k * 2 + 1];
    ab += w * b_in[k];
  }
  wc[g * 2] = a0; wc[g * 2 + 1] = a1; bc[g] = bsum[g] + ab;
}

// out[h*B + b] = in[b*H + h]  (the reference's transpose-flatten shuffle)
__global__ void transpose_kernel(const float* __restrict__ in, float* __restrict__ out) {
  __shared__ float t[32][33];
  int hb = blockIdx.x * 32, bb = blockIdx.y * 32;
  #pragma unroll
  for (int i = 0; i < 4; ++i)
    t[threadIdx.y + i * 8][threadIdx.x] =
        in[(size_t)(bb + threadIdx.y + i * 8) * HDIM + hb + threadIdx.x];
  __syncthreads();
  #pragma unroll
  for (int i = 0; i < 4; ++i)
    out[(size_t)(hb + threadIdx.y + i * 8) * BATCH + bb + threadIdx.x] =
        t[threadIdx.x][threadIdx.y + i * 8];
}

extern "C" void kernel_launch(void* const* d_in, const int* in_sizes, int n_in,
                              void* d_out, int out_size, void* d_ws, size_t ws_size,
                              hipStream_t stream) {
  const float* obs   = (const float*)d_in[0];
  const float* pre   = (const float*)d_in[1];
  const float* h0    = (const float*)d_in[2];
  const float* c0    = (const float*)d_in[3];
  const float* c1    = (const float*)d_in[4];
  const float* W_in  = (const float*)d_in[5];
  const float* b_in  = (const float*)d_in[6];
  const float* Wih_o = (const float*)d_in[7];
  const float* Whh_o = (const float*)d_in[8];
  const float* b_o   = (const float*)d_in[9];
  const float* Wih_p = (const float*)d_in[10];
  const float* Whh_p = (const float*)d_in[11];
  const float* b_p   = (const float*)d_in[12];
  float* out = (float*)d_out;

  char* ws = (char*)d_ws;
  const size_t WHH_ELEMS = (size_t)GDIM * HDIM;   // 4M
  const size_t HB        = (size_t)BATCH * HDIM;  // 4M
  unsigned short* whh_o_bf = (unsigned short*)ws; ws += WHH_ELEMS * 2;
  unsigned short* whh_p_bf = (unsigned short*)ws; ws += WHH_ELEMS * 2;
  unsigned short* hbf0     = (unsigned short*)ws; ws += HB * 2;
  unsigned short* hbf1     = (unsigned short*)ws; ws += HB * 2;
  float* hflt = (float*)ws; ws += HB * 4;
  float* cbuf = (float*)ws; ws += HB * 4;
  float* wc_o = (float*)ws; ws += (size_t)GDIM * 2 * 4;
  float* bc_o = (float*)ws; ws += (size_t)GDIM * 4;
  float* wc_p = (float*)ws; ws += (size_t)GDIM * 2 * 4;
  float* bc_p = (float*)ws; ws += (size_t)GDIM * 4;

  const int CT = 256;
  cvt_bf16_kernel<<<(int)((WHH_ELEMS + CT - 1) / CT), CT, 0, stream>>>(Whh_o, whh_o_bf, (int)WHH_ELEMS);
  cvt_bf16_kernel<<<(int)((WHH_ELEMS + CT - 1) / CT), CT, 0, stream>>>(Whh_p, whh_p_bf, (int)WHH_ELEMS);
  cvt_bf16_kernel<<<(int)((HB + CT - 1) / CT), CT, 0, stream>>>(h0, hbf0, (int)HB);
  compose_kernel<<<GDIM / CT, CT, 0, stream>>>(Wih_o, W_in, b_in, b_o, wc_o, bc_o);
  compose_kernel<<<GDIM / CT, CT, 0, stream>>>(Wih_p, W_in, b_in, b_p, wc_p, bc_p);
  hipMemcpyAsync(cbuf, c0, HB * 4, hipMemcpyDeviceToDevice, stream);

  dim3 sgrid(BATCH / 128, HDIM / 64);
  unsigned short* hb[2] = { hbf0, hbf1 };
  int cur = 0;
  for (int t = 0; t < T_OBS; ++t) {
    lstm_step_kernel<<<sgrid, 256, 0, stream>>>(hb[cur], hb[cur ^ 1], hflt, cbuf,
                                                whh_o_bf, wc_o, bc_o,
                                                obs + (size_t)t * BATCH * 2);
    cur ^= 1;
  }
  transpose_kernel<<<dim3(HDIM / 32, BATCH / 32), dim3(32, 8), 0, stream>>>(hflt, out);

  hipMemcpyAsync(cbuf, c1, HB * 4, hipMemcpyDeviceToDevice, stream);
  for (int t = 0; t < T_PRE; ++t) {
    lstm_step_kernel<<<sgrid, 256, 0, stream>>>(hb[cur], hb[cur ^ 1], hflt, cbuf,
                                                whh_p_bf, wc_p, bc_p,
                                                pre + (size_t)t * BATCH * 2);
    cur ^= 1;
  }
  transpose_kernel<<<dim3(HDIM / 32, BATCH / 32), dim3(32, 8), 0, stream>>>(hflt, out + HB);

  (void)in_sizes; (void)n_in; (void)out_size; (void)ws_size;
}
